// MiniGRU_parallel_61469571940855
// MI455X (gfx1250) — matrix operand order repaired and checked by reading, not compiled, not run on hardware
//
#include <hip/hip_runtime.h>
#include <hip/hip_bf16.h>

// ---------------------------------------------------------------------------
// MiniGRU parallel on gfx1250:
//   1) one-shot fp32 -> bf16 hi/lo split of x, W_z, W_h     (bandwidth pass)
//   2) fused dual-GEMM via v_wmma_f32_16x16x32_bf16, A tile staged in LDS
//      with async global->LDS double buffering, + log-space epilogue
//   3) chunked associative scan (aggregate / scan / apply)  (bandwidth passes)
// ---------------------------------------------------------------------------

typedef __attribute__((ext_vector_type(16))) __bf16 v16bf;
typedef __attribute__((ext_vector_type(8)))  __bf16 v8bf;
typedef __attribute__((ext_vector_type(8)))  float  v8f;
typedef __attribute__((ext_vector_type(4)))  int    v4i;

constexpr int S_LEN = 8192;
constexpr int K_DIM = 1024;
constexpr int H_DIM = 1024;

constexpr int NC = 64;    // scan chunks
constexpr int CH = 128;   // chunk length (NC*CH == S_LEN)

constexpr int AS_STRIDE = 40;   // LDS row stride in halfs (80B: 16B-aligned, bank-conflict-free)

// ---------------- async global->LDS (guarded; sync fallback) ---------------
#if defined(__has_builtin)
#  if __has_builtin(__builtin_amdgcn_global_load_async_to_lds_b128)
#    define HAVE_ASYNC_LDS 1
#  endif
#endif
#ifndef HAVE_ASYNC_LDS
#  define HAVE_ASYNC_LDS 0
#endif

__device__ __forceinline__ void stage_b128(const __bf16* gsrc, __bf16* ldst) {
#if HAVE_ASYNC_LDS
  __builtin_amdgcn_global_load_async_to_lds_b128(
      (__attribute__((address_space(1))) v4i*)gsrc,
      (__attribute__((address_space(3))) v4i*)ldst, 0, 0);
#else
  v4i v = *(const v4i*)gsrc;
  *(v4i*)ldst = v;
#endif
}

__device__ __forceinline__ void wait_async0() {
#if HAVE_ASYNC_LDS
#  if __has_builtin(__builtin_amdgcn_s_wait_asynccnt)
  __builtin_amdgcn_s_wait_asynccnt(0);
#  else
  asm volatile("s_wait_asynccnt 0x0" ::: "memory");
#  endif
#endif
}

// -------------------------- math helpers -----------------------------------

__device__ __forceinline__ float softplus_f(float v) {
  return fmaxf(v, 0.0f) + log1pf(__expf(-fabsf(v)));
}

__device__ __forceinline__ float log_g_f(float v) {
  return (v >= 0.0f) ? __logf(v + 0.5f) : -softplus_f(-v);
}

__device__ __forceinline__ float lae(float a, float b) {
  float m = fmaxf(a, b);
  if (m == -__builtin_inff()) return -__builtin_inff();
  return m + log1pf(__expf(-fabsf(a - b)));
}

__device__ __forceinline__ v16bf combine88(v8bf a, v8bf b) {
  return __builtin_shufflevector(a, b, 0,1,2,3,4,5,6,7,8,9,10,11,12,13,14,15);
}

// ---------------- kernel 0: fp32 -> bf16 hi/lo split (8 elems/thread) ------
__global__ __launch_bounds__(256) void
cvt_hilo(const float* __restrict__ src, __bf16* __restrict__ hi, __bf16* __restrict__ lo)
{
  const size_t i = ((size_t)blockIdx.x * blockDim.x + threadIdx.x) * 8;
  float4 f0 = *(const float4*)(src + i);
  float4 f1 = *(const float4*)(src + i + 4);
  float f[8] = { f0.x, f0.y, f0.z, f0.w, f1.x, f1.y, f1.z, f1.w };
  v8bf h, l;
#pragma unroll
  for (int j = 0; j < 8; ++j) {
    __bf16 hb = (__bf16)f[j];
    h[j] = hb;
    l[j] = (__bf16)(f[j] - (float)hb);
  }
  *(v8bf*)(hi + i) = h;
  *(v8bf*)(lo + i) = l;
}

// -------------------- kernel A: dual GEMM + epilogue ------------------------
// grid = (H/64, S/128), block = 256 (8 waves). Wave tile: 64(M) x 16(N),
// 4 m-tiles of 16x16, both W_z and W_h outputs. A tile staged in LDS (async,
// double-buffered); B fragments loaded directly (no cross-wave reuse).
__global__ __launch_bounds__(256) void
minigru_gemm_epilogue(const __bf16* __restrict__ xhi, const __bf16* __restrict__ xlo,
                      const __bf16* __restrict__ wzh, const __bf16* __restrict__ wzl,
                      const __bf16* __restrict__ whh, const __bf16* __restrict__ whl,
                      const float* __restrict__ bz,  const float* __restrict__ bh,
                      float* __restrict__ logc,     // [S,H] -> ws
                      float* __restrict__ logzh)    // [S,H] -> d_out (temp)
{
  __shared__ __bf16 sA[2][2][128 * AS_STRIDE];   // [buf][hi/lo][row*stride + k]

  const int tid    = threadIdx.x;
  const int lane   = tid & 31;
  const int wid    = tid >> 5;
  const int wave_m = wid & 1;        // 2 waves along M
  const int wave_n = wid >> 1;       // 4 waves along N
  const int m0     = blockIdx.y * 128;
  const int n_base = blockIdx.x * 64 + wave_n * 16;

  const int ml    = lane & 15;
  const int ksel  = lane >> 4;
  const int n_col = n_base + ml;

  v8f accz[4], acch[4];
#pragma unroll
  for (int mt = 0; mt < 4; ++mt) { accz[mt] = (v8f)(0.0f); acch[mt] = (v8f)(0.0f); }

  const __bf16* wzh_p = wzh + (size_t)n_col * K_DIM;
  const __bf16* wzl_p = wzl + (size_t)n_col * K_DIM;
  const __bf16* whh_p = whh + (size_t)n_col * K_DIM;
  const __bf16* whl_p = whl + (size_t)n_col * K_DIM;

  // ---- stage A tile for kb: 128 rows x 32 halfs, hi+lo, b128 chunks -------
  auto stage_tile = [&](int buf, int kb) {
#pragma unroll
    for (int p = 0; p < 2; ++p) {
      const int idx = tid + p * 256;          // 0..511
      const int r   = idx >> 2;               // 0..127
      const int c   = (idx & 3) * 8;          // k chunk of 8 halfs
      const size_t g = (size_t)(m0 + r) * K_DIM + kb + c;
      const int    o = r * AS_STRIDE + c;
      stage_b128(xhi + g, &sA[buf][0][o]);
      stage_b128(xlo + g, &sA[buf][1][o]);
    }
  };

  stage_tile(0, 0);

  for (int kb = 0, it = 0; kb < K_DIM; kb += 32, ++it) {
    const int cur = it & 1;
    wait_async0();
    __syncthreads();
    if (kb + 32 < K_DIM) stage_tile(cur ^ 1, kb + 32);

    // ---- B fragments: lane = column n_col, 16 contiguous K at 16*ksel ----
    const int ko = kb + ksel * 16;
    v16bf bzhf = *(const v16bf*)(wzh_p + ko);
    v16bf bzlf = *(const v16bf*)(wzl_p + ko);
    v16bf bhhf = *(const v16bf*)(whh_p + ko);
    v16bf bhlf = *(const v16bf*)(whl_p + ko);
    if (kb + 32 < K_DIM) {
      __builtin_prefetch(wzh_p + ko + 32, 0, 0);
      __builtin_prefetch(whh_p + ko + 32, 0, 0);
    }

    // ---- A fragments from LDS; wmma hi*hi + hi*lo + lo*hi -----------------
#pragma unroll
    for (int mt = 0; mt < 4; ++mt) {
      const int row_l = wave_m * 64 + mt * 16 + ml;
      const __bf16* ph = &sA[cur][0][row_l * AS_STRIDE];
      const __bf16* pl = &sA[cur][1][row_l * AS_STRIDE];
      v16bf ah = combine88(*(const v8bf*)(ph + ksel * 16), *(const v8bf*)(ph + ksel * 16 + 8));
      v16bf al = combine88(*(const v8bf*)(pl + ksel * 16), *(const v8bf*)(pl + ksel * 16 + 8));

      accz[mt] = __builtin_amdgcn_wmma_f32_16x16x32_bf16(false, ah, false, bzhf,
                                                         (short)0, accz[mt], false, false);
      accz[mt] = __builtin_amdgcn_wmma_f32_16x16x32_bf16(false, ah, false, bzlf,
                                                         (short)0, accz[mt], false, false);
      accz[mt] = __builtin_amdgcn_wmma_f32_16x16x32_bf16(false, al, false, bzhf,
                                                         (short)0, accz[mt], false, false);

      acch[mt] = __builtin_amdgcn_wmma_f32_16x16x32_bf16(false, ah, false, bhhf,
                                                         (short)0, acch[mt], false, false);
      acch[mt] = __builtin_amdgcn_wmma_f32_16x16x32_bf16(false, ah, false, bhlf,
                                                         (short)0, acch[mt], false, false);
      acch[mt] = __builtin_amdgcn_wmma_f32_16x16x32_bf16(false, al, false, bhhf,
                                                         (short)0, acch[mt], false, false);
    }
  }

  // ---- epilogue -----------------------------------------------------------
  const float bzc = bz[n_col];
  const float bhc = bh[n_col];
#pragma unroll
  for (int mt = 0; mt < 4; ++mt) {
#pragma unroll
    for (int r = 0; r < 8; ++r) {
      const int srow = m0 + wave_m * 64 + mt * 16 + ksel * 8 + r;   // C: M = r + 8*ksel
      float kv = accz[mt][r] + bzc;
      float ph = acch[mt][r] + bhc;
      float lz = -softplus_f(-kv);          // log(sigmoid(k))
      float lc = -softplus_f(kv);           // log(1 - sigmoid(k))
      float lt = log_g_f(ph);               // log_g(x W_h + b_h)
      const size_t o = (size_t)srow * H_DIM + n_col;
      logc[o]  = lc;
      logzh[o] = lz + lt;
    }
  }
}

// -------------------- kernel B: per-chunk aggregates ------------------------
__global__ __launch_bounds__(256) void
minigru_chunk_agg(const float* __restrict__ logc, const float* __restrict__ logzh,
                  float* __restrict__ aggC, float* __restrict__ aggD)
{
  const int h     = blockIdx.x * blockDim.x + threadIdx.x;
  const int chunk = blockIdx.y;
  float C = 0.0f;
  float D = -__builtin_inff();
  const size_t base = (size_t)chunk * CH * H_DIM + h;
  for (int i = 0; i < CH; ++i) {
    const size_t o = base + (size_t)i * H_DIM;
    float c = logc[o];
    float d = logzh[o];
    D = lae(c + D, d);
    C += c;
  }
  aggC[(size_t)chunk * H_DIM + h] = C;
  aggD[(size_t)chunk * H_DIM + h] = D;
}

// -------------------- kernel C: scan chunk aggregates -----------------------
__global__ __launch_bounds__(256) void
minigru_chunk_scan(const float* __restrict__ h0,
                   const float* __restrict__ aggC, const float* __restrict__ aggD,
                   float* __restrict__ pref)
{
  const int h = blockIdx.x * blockDim.x + threadIdx.x;
  float x = log_g_f(h0[h]);
  for (int c = 0; c < NC; ++c) {
    const size_t o = (size_t)c * H_DIM + h;
    pref[o] = x;
    x = lae(aggC[o] + x, aggD[o]);
  }
}

// -------------------- kernel D: apply prefixes + exp ------------------------
__global__ __launch_bounds__(256) void
minigru_apply(const float* __restrict__ logc, const float* __restrict__ pref,
              float* __restrict__ out)       // out holds logzh on entry
{
  const int h     = blockIdx.x * blockDim.x + threadIdx.x;
  const int chunk = blockIdx.y;
  float x = pref[(size_t)chunk * H_DIM + h];
  const size_t base = (size_t)chunk * CH * H_DIM + h;
  for (int i = 0; i < CH; ++i) {
    const size_t o = base + (size_t)i * H_DIM;
    float c = logc[o];
    float d = out[o];
    x = lae(c + x, d);
    out[o] = __expf(x);
  }
}

// ---------------------------------------------------------------------------

extern "C" void kernel_launch(void* const* d_in, const int* in_sizes, int n_in,
                              void* d_out, int out_size, void* d_ws, size_t ws_size,
                              hipStream_t stream) {
  (void)in_sizes; (void)n_in; (void)out_size; (void)ws_size;

  const float* x  = (const float*)d_in[0];
  const float* h0 = (const float*)d_in[1];
  const float* Wz = (const float*)d_in[2];
  const float* bz = (const float*)d_in[3];
  const float* Wh = (const float*)d_in[4];
  const float* bh = (const float*)d_in[5];
  float* out = (float*)d_out;

  // workspace layout (bytes): logc | x_hi | x_lo | wz_hi | wz_lo | wh_hi | wh_lo | aggC | aggD | pref
  char* w = (char*)d_ws;
  float* logc = (float*)w;           w += (size_t)S_LEN * H_DIM * sizeof(float);
  __bf16* xhi = (__bf16*)w;          w += (size_t)S_LEN * K_DIM * sizeof(__bf16);
  __bf16* xlo = (__bf16*)w;          w += (size_t)S_LEN * K_DIM * sizeof(__bf16);
  __bf16* wzh = (__bf16*)w;          w += (size_t)H_DIM * K_DIM * sizeof(__bf16);
  __bf16* wzl = (__bf16*)w;          w += (size_t)H_DIM * K_DIM * sizeof(__bf16);
  __bf16* whh = (__bf16*)w;          w += (size_t)H_DIM * K_DIM * sizeof(__bf16);
  __bf16* whl = (__bf16*)w;          w += (size_t)H_DIM * K_DIM * sizeof(__bf16);
  float* aggC = (float*)w;           w += (size_t)NC * H_DIM * sizeof(float);
  float* aggD = (float*)w;           w += (size_t)NC * H_DIM * sizeof(float);
  float* pref = (float*)w;

  // 0: precision split (removes all conversion VALU from the GEMM hot loop)
  cvt_hilo<<<(S_LEN * K_DIM) / 2048, 256, 0, stream>>>(x,  xhi, xlo);
  cvt_hilo<<<(H_DIM * K_DIM) / 2048, 256, 0, stream>>>(Wz, wzh, wzl);
  cvt_hilo<<<(H_DIM * K_DIM) / 2048, 256, 0, stream>>>(Wh, whh, whl);

  // A: fused GEMMs + elementwise epilogue (logzh staged in d_out)
  {
    dim3 grid(H_DIM / 64, S_LEN / 128);
    minigru_gemm_epilogue<<<grid, 256, 0, stream>>>(xhi, xlo, wzh, wzl, whh, whl,
                                                    bz, bh, logc, out);
  }
  // B: chunk aggregates
  {
    dim3 grid(H_DIM / 256, NC);
    minigru_chunk_agg<<<grid, 256, 0, stream>>>(logc, out, aggC, aggD);
  }
  // C: scan over chunk aggregates
  minigru_chunk_scan<<<dim3(H_DIM / 256), 256, 0, stream>>>(h0, aggC, aggD, pref);
  // D: apply prefixes, exponentiate in place
  {
    dim3 grid(H_DIM / 256, NC);
    minigru_apply<<<grid, 256, 0, stream>>>(logc, pref, out);
  }
}
